// DenseCorr2d_full_68118181314795
// MI455X (gfx1250) — matrix operand order repaired and checked by reading, not compiled, not run on hardware
//
#include <hip/hip_runtime.h>

// ---------------- CDNA5 WMMA types ----------------
typedef __bf16 bf16_t;
typedef __attribute__((ext_vector_type(16))) __bf16 v16bf;
typedef __attribute__((ext_vector_type(8)))  __bf16 v8bf;
typedef __attribute__((ext_vector_type(8)))  float  v8f;
typedef __attribute__((ext_vector_type(4)))  float  v4f;

// ---------------- Problem constants ----------------
#define B_   8
#define CT   16
#define CM   16
#define HM   128
#define WM   128
#define COUT 64
#define CIN  256          // CM*CT
#define K2   2304         // CIN*9

// tile: 16 wide x 8 tall outputs per workgroup
#define TX   16
#define TY   8
#define CX   18           // corr tile width  (TX + 2 halo)
#define CY   10           // corr tile height (TY + 2 halo)
#define NPIX 180          // CX*CY
#define INX  33           // CX + 15
#define INY  25           // CY + 15

__device__ __forceinline__ int iclamp(int v, int lo, int hi) {
  return v < lo ? lo : (v > hi ? hi : v);
}

// ---- prep: W [64][256][3][3] f32  ->  Wt [64][9][256] bf16 (k = r*256 + ci) ----
__global__ __launch_bounds__(256)
void prep_w(const float* __restrict__ Wg, bf16_t* __restrict__ Wt) {
  int i = blockIdx.x * 256 + threadIdx.x;
  if (i < COUT * K2) {
    int co  = i / K2;
    int rem = i - co * K2;
    int ci  = rem / 9;
    int r   = rem - ci * 9;
    Wt[co * K2 + r * 256 + ci] = (bf16_t)Wg[i];
  }
}

__global__ __launch_bounds__(256)
void densecorr_merge_fused(const float* __restrict__ tmpl,
                           const float* __restrict__ tomatch,
                           const bf16_t* __restrict__ Wt,   // pre-swizzled [64][9][256]
                           const float* __restrict__ bias,
                           float* __restrict__ out)
{
  extern __shared__ char smem_raw[];
  bf16_t* sB1f  = (bf16_t*)smem_raw;        // [chunk=8][lane=32][u=16] template frags
  bf16_t* sIn   = sB1f + 8 * 32 * 16;       // [CM][INY][INX]
  bf16_t* sCorr = sIn + CM * INY * INX;     // [NPIX][CIN] pixel-major, ci contiguous

  const int tid  = threadIdx.x;
  const int lane = tid & 31;
  const int wave = tid >> 5;
  const int mrow = lane & 15;
  const bool hi  = lane >= 16;
  const int hi8  = hi ? 8 : 0;

  const int x0 = blockIdx.x * TX;
  const int y0 = blockIdx.y * TY;
  const int b  = blockIdx.z;

  // ---------- Phase A: stage template (swizzled) + input tile into LDS ----------
  const float* tp = tmpl + b * (CT * 256);
  for (int i = tid; i < 8 * 32 * 16; i += 256) {
    int c   = i >> 9;
    int rem = i & 511;
    int ln  = rem >> 4;
    int u   = rem & 15;
    int ct  = ln & 15;
    int kk  = ((ln & 16) ? 8 : 0) + (u < 8 ? u : u + 8);
    sB1f[i] = (bf16_t)tp[ct * 256 + (c * 32 + kk)];
  }
  const float* tmb = tomatch + (size_t)b * CM * HM * WM;
  for (int i = tid; i < CM * INY * INX; i += 256) {
    int cm = i / (INY * INX);
    int r  = i - cm * (INY * INX);
    int iy = r / INX, ix = r - (r / INX) * INX;
    int gy = iclamp(y0 - 1 + iy, 0, HM - 1);      // edge replication pad
    int gx = iclamp(x0 - 1 + ix, 0, WM - 1);
    sIn[i] = (bf16_t)tmb[cm * (HM * WM) + gy * WM + gx];
  }
  __syncthreads();

  // ---------- Phase B: corr GEMM (M=pixels, N=ct, K=256) ----------
  v16bf bfrag[8];
  #pragma unroll
  for (int c = 0; c < 8; ++c)
    bfrag[c] = *(const v16bf*)(sB1f + c * 512 + lane * 16);

  for (int job = wave; job < 16 * 12; job += 8) {
    int cm = job / 12;
    int pt = job - cm * 12;
    int p  = pt * 16 + mrow;
    int pc = p < NPIX ? p : 0;
    int py = pc / CX, px = pc - (pc / CX) * CX;
    const bf16_t* inb = sIn + cm * (INY * INX) + py * INX + px;

    // Issue ALL A-chunk loads first (immediate DS offsets), then 8 WMMAs:
    // compiler emits descending s_wait_dscnt so loads overlap matrix issue.
    v16bf av[8];
    #pragma unroll
    for (int c = 0; c < 8; ++c) {
      int kc = c * 32, ka = hi8;
      #pragma unroll
      for (int u = 0; u < 8; ++u) {
        int k0 = kc + ka + u;
        int k1 = kc + ka + 16 + u;
        av[c][u]     = inb[(k0 >> 4) * INX + (k0 & 15)];
        av[c][8 + u] = inb[(k1 >> 4) * INX + (k1 & 15)];
      }
    }
    v8f acc = {0.f, 0.f, 0.f, 0.f, 0.f, 0.f, 0.f, 0.f};
    #pragma unroll
    for (int c = 0; c < 8; ++c)
      acc = __builtin_amdgcn_wmma_f32_16x16x32_bf16(false, av[c], false, bfrag[c],
                                                    (short)0, acc, false, false);

    // scatter D to sCorr[pixel][ci]; zero halo outside [0,128) (SAME zero-pad).
    // Incremental qx/qy carry instead of per-element division by 18.
    int ci  = cm * 16 + mrow;                 // N = ct
    int pp0 = pt * 16 + hi8;
    int qy  = pp0 / CX;
    int qx  = pp0 - qy * CX;
    int gy  = y0 - 1 + qy;
    int gx  = x0 - 1 + qx;
    bf16_t* sc = sCorr + pp0 * CIN + ci;      // advance by 256 el per r (imm offsets)
    #pragma unroll
    for (int r = 0; r < 8; ++r) {
      if (pp0 + r < NPIX) {
        float val = ((unsigned)gy < (unsigned)HM && (unsigned)gx < (unsigned)WM)
                        ? acc[r] : 0.0f;
        sc[r * CIN] = (bf16_t)val;
      }
      ++qx; ++gx;
      if (qx == CX) { qx = 0; gx = x0 - 1; ++gy; }
    }
  }
  __syncthreads();

  // ---------- Phase C: merge conv GEMM (M=x, N=co, K=9*256, k=r*256+ci) ----------
  for (int job = wave; job < 8 * 4; job += 8) {
    int t  = job >> 2;                        // output row within tile
    int nt = job & 3;                         // co tile
    int co = nt * 16 + mrow;                  // N = co for this lane
    float bv = bias[co];
    v8f acc = {bv, bv, bv, bv, bv, bv, bv, bv};

    // base pointers; all 72 chunks become base + (folded) immediate offset
    const bf16_t* pA = sCorr + (t * CX + mrow) * CIN + (hi ? 16 : 0);
    const bf16_t* pB = Wt + (size_t)co * K2 + (hi ? 16 : 0);

    // depth-2 software pipeline: load chunk k+1 while WMMA-ing chunk k.
    // (plain ints: fully-unrolled loop makes these compile-time constants)
    #define LOAD_A(kn, dst)                                                   \
      {                                                                       \
        int r_  = (kn) >> 3;                                                  \
        int c8_ = (kn) & 7;                                                   \
        int ky_ = r_ / 3, kx_ = r_ - (r_ / 3) * 3;                            \
        int off = (ky_ * CX + kx_) * CIN + c8_ * 32;                          \
        v8bf alo = *(const v8bf*)(pA + off);                                  \
        v8bf ahi = *(const v8bf*)(pA + off + 8);                              \
        dst = __builtin_shufflevector(alo, ahi, 0, 1, 2, 3, 4, 5, 6, 7,       \
                                      8, 9, 10, 11, 12, 13, 14, 15);          \
      }
    #define LOAD_B(kn, dst)                                                   \
      {                                                                       \
        int off = ((kn) >> 3) * 256 + ((kn) & 7) * 32;                        \
        dst = *(const v16bf*)(pB + off);                                      \
      }

    v16bf aP, bP;
    LOAD_A(0, aP);
    LOAD_B(0, bP);
    #pragma unroll
    for (int kidx = 0; kidx < 72; ++kidx) {
      if (kidx < 71) {
        v16bf aN, bN;
        LOAD_A(kidx + 1, aN);
        LOAD_B(kidx + 1, bN);
        acc = __builtin_amdgcn_wmma_f32_16x16x32_bf16(false, aP, false, bP,
                                                      (short)0, acc, false, false);
        aP = aN; bP = bN;
      } else {
        acc = __builtin_amdgcn_wmma_f32_16x16x32_bf16(false, aP, false, bP,
                                                      (short)0, acc, false, false);
      }
    }
    #undef LOAD_A
    #undef LOAD_B

    // write D: lane N=co, VGPR rr -> x offset; contiguous 32B per lane
    float* op = out + ((size_t)b * COUT + co) * (HM * WM) + (y0 + t) * WM + x0 + hi8;
    v4f lo = {acc[0], acc[1], acc[2], acc[3]};
    v4f hi4 = {acc[4], acc[5], acc[6], acc[7]};
    *(v4f*)(op)     = lo;
    *(v4f*)(op + 4) = hi4;
  }
}

extern "C" void kernel_launch(void* const* d_in, const int* in_sizes, int n_in,
                              void* d_out, int out_size, void* d_ws, size_t ws_size,
                              hipStream_t stream) {
  (void)in_sizes; (void)n_in; (void)out_size; (void)ws_size;
  const float* tmpl    = (const float*)d_in[0];   // [8,16,16,16]
  const float* tomatch = (const float*)d_in[1];   // [8,16,128,128]
  const float* Wg      = (const float*)d_in[2];   // [64,256,3,3]
  const float* bias    = (const float*)d_in[3];   // [64]
  float* out  = (float*)d_out;                    // [8,64,128,128]
  bf16_t* Wt  = (bf16_t*)d_ws;                    // 294912 B scratch

  prep_w<<<(COUT * K2 + 255) / 256, 256, 0, stream>>>(Wg, Wt);

  constexpr size_t SMEM =
      (size_t)(8 * 32 * 16 + CM * INY * INX + NPIX * CIN) * sizeof(bf16_t); // 126752 B
  static_assert(SMEM <= 320 * 1024, "fits WGP LDS");
  (void)hipFuncSetAttribute((const void*)densecorr_merge_fused,
                            hipFuncAttributeMaxDynamicSharedMemorySize, (int)SMEM);

  dim3 grid(WM / TX, HM / TY, B_);   // (8,16,8)
  dim3 block(256);                   // 8 wave32
  densecorr_merge_fused<<<grid, block, SMEM, stream>>>(tmpl, tomatch, Wt, bias, out);
}
